// InversePieceWiseLinearCoupling_36833639530645
// MI455X (gfx1250) — matrix operand order repaired and checked by reading, not compiled, not run on hardware
//
#include <hip/hip_runtime.h>

// ---------------------------------------------------------------------------
// InversePieceWiseLinearCoupling for MI455X (gfx1250, wave32, WMMA + TDM)
//
//   K0: w2 (1024x1024 f32, [k][n]) -> w2t (f16, [n][k])      (B-operand layout)
//   K0: w3 (1024x512  f32, [k][n]) -> w3t (f16, [n][k])
//   K1: h1 = relu(yA @ w1 + b1)            -> f16 [B][1024]  (K=8, VALU)
//   K2: h2 = relu(h1 @ w2 + b2)            -> f16 [B][1024]  (WMMA, TDM staging)
//   K3: q  = softplus(h2 @ w3 + b3) + 1e-6 -> f32 [B][512]   (WMMA, TDM staging)
//   K4: per-row 64-bin piecewise-linear inversion + Jacobian -> out [B][17]
// ---------------------------------------------------------------------------

typedef __attribute__((ext_vector_type(16))) _Float16     v16h;
typedef __attribute__((ext_vector_type(8)))  _Float16     v8h;
typedef __attribute__((ext_vector_type(8)))  float        v8f;
typedef __attribute__((ext_vector_type(4)))  unsigned int u32x4;
typedef __attribute__((ext_vector_type(8)))  int          i32x8;
typedef __attribute__((ext_vector_type(4)))  int          i32x4;

union Frag16 { v16h v; v8h h[2]; };

#define B_ROWS 65536
#define HID    1024
#define QDIM   512
#define NBINS  64
#define BK     64    // K-chunk per TDM tile (halves)
#define LPITCH 72    // LDS row pitch in halves: 64 data + 8 pad (TDM pad_interval=4/amount=3)

// ---------------------------------------------------------------------------
// TDM: DMA a 64-row x 64-half f16 tile (row stride = stride_elems halves in
// global) into LDS at lds_byte_off, inserting a 4-DWORD pad after each
// 32-DWORD row so the LDS pitch is 72 halves (bank-spread, 16B aligned).
// Descriptor per CDNA5 ISA section 8 (D# groups 0/1; groups 2/3 zero = 2D).
// This toolchain exposes the 6-arg builtin (clang-23 / therock-10.0 lane).
// ---------------------------------------------------------------------------
__device__ __forceinline__ void tdm_load_tile64x64(unsigned lds_byte_off,
                                                   const _Float16* gsrc,
                                                   unsigned stride_elems) {
    unsigned long long ga = (unsigned long long)(size_t)gsrc;
    u32x4 g0 = { 1u,                                   // count=1, gather off
                 lds_byte_off,                          // lds_addr (bytes)
                 (unsigned)ga,                          // global_addr[31:0]
                 (unsigned)((ga >> 32) & 0x01FFFFFFu)   // global_addr[56:32]
                   | (2u << 30) };                      // type=2 ("image")
    i32x8 g1;
    g1[0] = (1 << 16)      // data_size = 1 -> 2 bytes
          | (1 << 20)      // pad_enable
          | (4 << 22)      // pad_interval = 4 -> every 32 DWORDs (one 64-half row)
          | (3 << 25);     // pad_amount   = 3 -> 4 DWORDs (8 halves)
    g1[1] = (int)((stride_elems & 0xFFFFu) << 16);          // tensor_dim0[15:0]
    g1[2] = (int)((stride_elems >> 16) & 0xFFFFu);          // dim0[31:16] | dim1[15:0]=0
    g1[3] = (int)(0x10u | (64u << 16));                     // dim1=1<<20 | tile_dim0=64
    g1[4] = 64;                                             // tile_dim1=64, tile_dim2=0
    g1[5] = (int)stride_elems;                              // tensor_dim0_stride[31:0]
    g1[6] = 0;                                              // stride hi | dim1_stride lo
    g1[7] = 0;
    i32x4 z4 = {0, 0, 0, 0};
    i32x8 z8 = {0, 0, 0, 0, 0, 0, 0, 0};
    __builtin_amdgcn_tensor_load_to_lds(g0, g1, z4, z4, z8, 0);
}

// ---------------------------------------------------------------------------
// K0: transpose + f32->f16 convert:  src[K][N] -> dst[N][K]   (K = 1024)
// ---------------------------------------------------------------------------
__global__ void conv_transpose_f16(const float* __restrict__ src,
                                   _Float16* __restrict__ dst, int N) {
    int gid = blockIdx.x * blockDim.x + threadIdx.x;
    int total = HID * N;
    if (gid >= total) return;
    int k = gid / N;
    int n = gid - k * N;
    dst[(size_t)n * HID + k] = (_Float16)src[(size_t)k * N + n];
}

// ---------------------------------------------------------------------------
// K1: h1 = relu(yA @ w1 + b1).  One block per row; K=8 so plain FMA.
// ---------------------------------------------------------------------------
__global__ __launch_bounds__(256) void gemm1_relu(const float* __restrict__ y,
                                                  const float* __restrict__ w1,
                                                  const float* __restrict__ b1,
                                                  _Float16* __restrict__ h1) {
    const int row = blockIdx.x;
    const float* yr = y + (size_t)row * 17;
    float a[8];
#pragma unroll
    for (int k = 0; k < 8; ++k) a[k] = yr[k];
#pragma unroll
    for (int it = 0; it < 4; ++it) {
        int j = threadIdx.x + it * 256;
        float acc = b1[j];
#pragma unroll
        for (int k = 0; k < 8; ++k) acc = fmaf(a[k], w1[k * HID + j], acc);
        h1[(size_t)row * HID + j] = (_Float16)fmaxf(acc, 0.0f);
    }
}

// ---------------------------------------------------------------------------
// WMMA GEMM core: 64x64 C tile per block. A = act f16 [M][1024] row-major,
// B = wt f16 [N][1024] (transposed weights -> K contiguous, matches B layout).
// 8 waves = 4 M-blocks x 2 N-halves; TDM double-buffered K panels (BK=64).
// ---------------------------------------------------------------------------
template <int NOUT, int ACT> // ACT 0: relu->f16 ; ACT 1: softplus+1e-6 ->f32
__device__ __forceinline__ void gemm_wmma_tile(const _Float16* __restrict__ act,
                                               const _Float16* __restrict__ wt,
                                               const float* __restrict__ bias,
                                               _Float16* __restrict__ out16,
                                               float* __restrict__ out32) {
    __shared__ __align__(16) _Float16 sA[2][64 * LPITCH];
    __shared__ __align__(16) _Float16 sB[2][64 * LPITCH];

    const int tid  = threadIdx.x;
    const int lane = tid & 31;
    const int wave = tid >> 5;
    const int mb   = wave & 3;   // M block (16 rows)
    const int nb2  = wave >> 2;  // N half (32 cols)
    const size_t m0 = (size_t)blockIdx.y * 64;
    const int    n0 = blockIdx.x * 64;

    v8f acc0 = {0.f, 0.f, 0.f, 0.f, 0.f, 0.f, 0.f, 0.f};
    v8f acc1 = {0.f, 0.f, 0.f, 0.f, 0.f, 0.f, 0.f, 0.f};

    const unsigned sA0 = (unsigned)(size_t)&sA[0][0];
    const unsigned sA1 = (unsigned)(size_t)&sA[1][0];
    const unsigned sB0 = (unsigned)(size_t)&sB[0][0];
    const unsigned sB1 = (unsigned)(size_t)&sB[1][0];

    if (wave == 0) {  // prime the pipeline: chunk 0 -> buffer 0
        tdm_load_tile64x64(sA0, act + m0 * HID, HID);
        tdm_load_tile64x64(sB0, wt + (size_t)n0 * HID, HID);
    }

    const int NCHUNK = HID / BK;  // 16
    for (int c = 0; c < NCHUNK; ++c) {
        const int buf = c & 1;
        __builtin_amdgcn_s_wait_tensorcnt((short)0);  // chunk c landed (no-op for waves>0)
        __syncthreads();                               // publish buffer + retire prev reads

        if (wave == 0 && c + 1 < NCHUNK) {             // DMA chunk c+1 into other buffer
            const int kt = (c + 1) * BK;
            tdm_load_tile64x64(buf ? sA0 : sA1, act + m0 * HID + kt, HID);
            tdm_load_tile64x64(buf ? sB0 : sB1, wt + (size_t)n0 * HID + kt, HID);
        }

        const _Float16* A = sA[buf];
        const _Float16* Bp = sB[buf];
#pragma unroll
        for (int ks = 0; ks < 2; ++ks) {  // 2 WMMA K-steps of 32 per chunk
            // A frag: lane<16 holds K 0..15, lane>=16 holds K 16..31
            Frag16 a;
            int rl  = mb * 16 + (lane & 15);
            int klo = ks * 32 + ((lane & 16) ? 16 : 0);
            a.h[0] = *(const v8h*)&A[rl * LPITCH + klo];
            a.h[1] = *(const v8h*)&A[rl * LPITCH + klo + 8];

            // B frags: lane<16 -> K 0..15 of col N=lane ; lane>=16 -> K 16..31
            int kb  = ks * 32 + ((lane & 16) ? 16 : 0);
            int cl0 = nb2 * 32 + (lane & 15);
            Frag16 b0, b1;
            b0.h[0] = *(const v8h*)&Bp[cl0 * LPITCH + kb];
            b0.h[1] = *(const v8h*)&Bp[cl0 * LPITCH + kb + 8];
            b1.h[0] = *(const v8h*)&Bp[(cl0 + 16) * LPITCH + kb];
            b1.h[1] = *(const v8h*)&Bp[(cl0 + 16) * LPITCH + kb + 8];

            acc0 = __builtin_amdgcn_wmma_f32_16x16x32_f16(false, a.v, false, b0.v,
                                                          (short)0, acc0, false, false);
            acc1 = __builtin_amdgcn_wmma_f32_16x16x32_f16(false, a.v, false, b1.v,
                                                          (short)0, acc1, false, false);
        }
        __syncthreads();  // all reads of buf done before TDM may overwrite it
    }

    // epilogue: C/D layout -> lane<16: N=lane, M=v ; lane>=16: N=lane-16, M=v+8
    const int nA = n0 + nb2 * 32 + (lane & 15);
    const int nB = nA + 16;
    const float bA = bias[nA];
    const float bB = bias[nB];
    const int mbase = mb * 16 + ((lane & 16) ? 8 : 0);
#pragma unroll
    for (int v = 0; v < 8; ++v) {
        size_t r = m0 + mbase + v;
        float xA = acc0[v] + bA;
        float xB = acc1[v] + bB;
        if (ACT == 0) {
            out16[r * NOUT + nA] = (_Float16)fmaxf(xA, 0.0f);
            out16[r * NOUT + nB] = (_Float16)fmaxf(xB, 0.0f);
        } else {
            float sA2 = ((xA > 20.f) ? xA : log1pf(__expf(xA))) + 1e-6f;
            float sB2 = ((xB > 20.f) ? xB : log1pf(__expf(xB))) + 1e-6f;
            out32[r * NOUT + nA] = sA2;
            out32[r * NOUT + nB] = sB2;
        }
    }
}

__global__ __launch_bounds__(256) void gemm2_relu(const _Float16* __restrict__ h1,
                                                  const _Float16* __restrict__ w2t,
                                                  const float* __restrict__ b2,
                                                  _Float16* __restrict__ h2) {
    gemm_wmma_tile<HID, 0>(h1, w2t, b2, h2, nullptr);
}

__global__ __launch_bounds__(256) void gemm3_softplus(const _Float16* __restrict__ h2,
                                                      const _Float16* __restrict__ w3t,
                                                      const float* __restrict__ b3,
                                                      float* __restrict__ q) {
    gemm_wmma_tile<QDIM, 1>(h2, w3t, b3, nullptr, q);
}

// ---------------------------------------------------------------------------
// K4: per-(row,channel) inversion. gid -> row = gid/8, ch = gid%8.
// ---------------------------------------------------------------------------
__global__ __launch_bounds__(256) void invert_epilogue(const float* __restrict__ y,
                                                       const float* __restrict__ q,
                                                       float* __restrict__ out) {
    int gid = blockIdx.x * blockDim.x + threadIdx.x;
    int row = gid >> 3;
    int ch  = gid & 7;
    const float* qr = q + (size_t)row * QDIM + ch * NBINS;

    float qv[NBINS];
    float total = 0.0f;
#pragma unroll
    for (int i = 0; i < NBINS; i += 4) {
        float4 v4 = *(const float4*)(qr + i);
        qv[i] = v4.x; qv[i + 1] = v4.y; qv[i + 2] = v4.z; qv[i + 3] = v4.w;
        total += v4.x + v4.y + v4.z + v4.w;
    }
    const float inv = 1.0f / total;
    const float yB  = y[(size_t)row * 17 + 8 + ch];

    int   bin    = 0;
    float cum    = 0.0f;
    float offacc = 0.0f;
    float slopeq = qv[NBINS - 1];
    bool  found  = false;
#pragma unroll
    for (int i = 0; i < NBINS; ++i) {
        float c2 = cum + qv[i];
        bool  lt = (c2 * inv) < yB;
        bin += lt ? 1 : 0;
        offacc = lt ? c2 : offacc;
        slopeq = (!lt && !found) ? qv[i] : slopeq;
        found  = found || !lt;
        cum    = c2;
    }
    if (bin > NBINS - 1) bin = NBINS - 1;

    float offset = offacc * inv;
    float slope  = slopeq * inv * (float)NBINS;
    float xB     = (yB - offset) / slope + (float)bin * (1.0f / (float)NBINS);

    float pr = 1.0f / slope;
    pr *= __shfl_xor(pr, 1);
    pr *= __shfl_xor(pr, 2);
    pr *= __shfl_xor(pr, 4);

    out[(size_t)row * 17 + ch]     = y[(size_t)row * 17 + ch];
    out[(size_t)row * 17 + 8 + ch] = xB;
    if (ch == 0) out[(size_t)row * 17 + 16] = y[(size_t)row * 17 + 16] * pr;
}

// ---------------------------------------------------------------------------
extern "C" void kernel_launch(void* const* d_in, const int* in_sizes, int n_in,
                              void* d_out, int out_size, void* d_ws, size_t ws_size,
                              hipStream_t stream) {
    const float* y  = (const float*)d_in[0];
    const float* w1 = (const float*)d_in[1];
    const float* b1 = (const float*)d_in[2];
    const float* w2 = (const float*)d_in[3];
    const float* b2 = (const float*)d_in[4];
    const float* w3 = (const float*)d_in[5];
    const float* b3 = (const float*)d_in[6];
    float* out = (float*)d_out;

    char* ws = (char*)d_ws;
    _Float16* w2t = (_Float16*)(ws);                                   //  2 MB
    _Float16* w3t = (_Float16*)(ws + (size_t)HID * HID * 2);           //  1 MB
    _Float16* h1  = (_Float16*)(ws + (size_t)HID * (HID + QDIM) * 2);  // 128 MB
    _Float16* h2  = h1 + (size_t)B_ROWS * HID;                         // 128 MB
    float*    q   = (float*)(h2 + (size_t)B_ROWS * HID);               // 128 MB

    conv_transpose_f16<<<(HID * HID + 255) / 256, 256, 0, stream>>>(w2, w2t, HID);
    conv_transpose_f16<<<(HID * QDIM + 255) / 256, 256, 0, stream>>>(w3, w3t, QDIM);

    gemm1_relu<<<B_ROWS, 256, 0, stream>>>(y, w1, b1, h1);

    gemm2_relu<<<dim3(HID / 64, B_ROWS / 64), 256, 0, stream>>>(h1, w2t, b2, h2);
    gemm3_softplus<<<dim3(QDIM / 64, B_ROWS / 64), 256, 0, stream>>>(h2, w3t, b3, q);

    invert_epilogue<<<(B_ROWS * 8) / 256, 256, 0, stream>>>(y, q, out);
}
